// RNNModel_90211493085747
// MI455X (gfx1250) — matrix operand order repaired and checked by reading, not compiled, not run on hardware
//
#include <hip/hip_runtime.h>
#include <math.h>

#define T_DIM 256
#define B_DIM 128
#define I_DIM 512
#define H_DIM 1024
#define LDS_STRIDE 1032   // 1024 + 8 bf16 pad -> conflict-free ds_load_b128

typedef __attribute__((ext_vector_type(16))) __bf16 v16bf;
typedef __attribute__((ext_vector_type(8)))  __bf16 v8bf;
typedef __attribute__((ext_vector_type(4)))  __bf16 v4bf;
typedef __attribute__((ext_vector_type(8)))  float  v8f;
typedef __attribute__((ext_vector_type(4)))  float  v4f;

// ---------------- fragment helpers (CDNA5 WMMA 16x16x32 bf16 layouts) -------

// B-matrix fragment: two 8-bf16 chunks at p (K = 8g..) and p+16 (K = 16+8g..)
// (p already includes the per-lane row/kh offsets).
static __device__ __attribute__((always_inline)) v16bf ld_bfrag(const __bf16* p) {
    v8bf lo = *(const v8bf*)p;
    v8bf hi = *(const v8bf*)(p + 16);
    return __builtin_shufflevector(lo, hi, 0,1,2,3,4,5,6,7,8,9,10,11,12,13,14,15);
}

// A-matrix fragment: two 8-bf16 chunks at p (K = 8g..) and p+16 (K = 16+8g..).
static __device__ __attribute__((always_inline)) v16bf ld_afrag(const __bf16* p) {
    v8bf lo = *(const v8bf*)p;
    v8bf hi = *(const v8bf*)(p + 16);
    return __builtin_shufflevector(lo, hi, 0,1,2,3,4,5,6,7,8,9,10,11,12,13,14,15);
}

static __device__ __attribute__((always_inline)) v8f wmma_bf16(v16bf a, v16bf b, v8f c) {
    return __builtin_amdgcn_wmma_f32_16x16x32_bf16(false, a, false, b, (short)0, c,
                                                   false, false);
}

static __device__ __attribute__((always_inline)) float fast_tanh(float x) {
    // tanh(x) = 1 - 2/(exp(2x)+1); saturates correctly at +-1
    float e = __expf(2.0f * x);
    return 1.0f - 2.0f / (e + 1.0f);
}

// ---------------- elementwise f32 -> bf16 conversion ------------------------

__global__ void cvt_f32_to_bf16(const float* __restrict__ in,
                                __bf16* __restrict__ out, int n4) {
    int i = blockIdx.x * blockDim.x + threadIdx.x;
    if (i < n4) {
        v4f v = *(const v4f*)(in + (size_t)i * 4);
        v4bf o;
        o.x = (__bf16)v.x; o.y = (__bf16)v.y; o.z = (__bf16)v.z; o.w = (__bf16)v.w;
        *(v4bf*)(out + (size_t)i * 4) = o;
    }
}

// ---------------- xp = x @ W_ih^T + b_ih + b_hh  (M=32768, K=512, N=1024) ---
// One wave computes a 16(M) x 64(N) tile; 8 waves per block; 4096 blocks.

__global__ void xp_gemm(const __bf16* __restrict__ xbf,   // [M, K] row-major
                        const __bf16* __restrict__ wih,   // [N, K] row-major
                        const float* __restrict__ b_ih,
                        const float* __restrict__ b_hh,
                        float* __restrict__ xp)           // [M, N]
{
    int lane = threadIdx.x & 31;
    int wave = threadIdx.x >> 5;
    int gw   = blockIdx.x * 8 + wave;     // 0..32767
    int m0   = (gw >> 4) * 16;            // M tile base
    int n0   = (gw & 15) * 64;            // N tile base
    int g = lane >> 4, nl = lane & 15;

    const __bf16* arow = xbf + (size_t)(m0 + nl) * I_DIM + g * 8;

    v8f acc[4] = {};
    for (int k0 = 0; k0 < I_DIM; k0 += 32) {
        v16bf a = ld_afrag(arow + k0);
#pragma unroll
        for (int c = 0; c < 4; ++c) {
            const __bf16* bp = wih + (size_t)(n0 + c * 16 + nl) * I_DIM + k0 + g * 8;
            acc[c] = wmma_bf16(a, ld_bfrag(bp), acc[c]);
        }
    }
#pragma unroll
    for (int c = 0; c < 4; ++c) {
        int n = n0 + c * 16 + nl;
        float bias = b_ih[n] + b_hh[n];
#pragma unroll
        for (int r = 0; r < 8; ++r) {
            int m = m0 + r + g * 8;
            xp[(size_t)m * H_DIM + n] = acc[c][r] + bias;
        }
    }
}

// ---------------- persistent recurrence kernel ------------------------------
// 16 blocks x 256 threads (8 waves). Block b owns columns [64b, 64b+64) of h;
// wave w owns rows [16w, 16w+16). W_hh slice (64 rows x 1024) staged in LDS.
// Grid-wide sense barrier between timesteps.

static __device__ __attribute__((always_inline))
void grid_barrier(unsigned* count, unsigned* gen, unsigned nblocks) {
    __syncthreads();
    if (threadIdx.x == 0) {
        __threadfence();
        unsigned g = __atomic_load_n(gen, __ATOMIC_ACQUIRE);
        unsigned old = atomicAdd(count, 1u);
        if (old == nblocks - 1u) {
            __atomic_store_n(count, 0u, __ATOMIC_RELAXED);
            __threadfence();
            atomicAdd(gen, 1u);
        } else {
            while (__atomic_load_n(gen, __ATOMIC_ACQUIRE) == g) {
                __builtin_amdgcn_s_sleep(1);
            }
        }
        __threadfence();
    }
    __syncthreads();
}

__global__ void rnn_steps(const float* __restrict__ xp,    // [T*B, H] fp32
                          const __bf16* __restrict__ whh,  // [H, H] bf16 row-major
                          __bf16* __restrict__ hs,         // [T*B, H] bf16 out
                          unsigned* __restrict__ bar)
{
    extern __shared__ __bf16 lds_w[];   // 64 rows x LDS_STRIDE bf16 (~129 KB)

    int lane = threadIdx.x & 31;
    int wave = threadIdx.x >> 5;
    int n0   = blockIdx.x * 64;         // column strip for this block
    int m0   = wave * 16;               // row strip for this wave
    int g = lane >> 4, nl = lane & 15;

    // Stage this block's W_hh rows [n0, n0+64) into LDS (time-invariant).
    {
        const __bf16* wsrc = whh + (size_t)n0 * H_DIM;
        for (int idx = threadIdx.x; idx < 64 * (H_DIM / 8); idx += blockDim.x) {
            int r = idx >> 7;           // H_DIM/8 == 128 chunks per row
            int c = idx & 127;
            *(v8bf*)(lds_w + (size_t)r * LDS_STRIDE + c * 8) =
                *(const v8bf*)(wsrc + (size_t)r * H_DIM + c * 8);
        }
    }
    __syncthreads();

    for (int t = 0; t < T_DIM; ++t) {
        v8f acc[4] = {};
        if (t > 0) {
            const __bf16* hprev = hs + (size_t)(t - 1) * B_DIM * H_DIM;
            const __bf16* arow  = hprev + (size_t)(m0 + nl) * H_DIM + g * 8;
#pragma unroll 2
            for (int k0 = 0; k0 < H_DIM; k0 += 32) {
                v16bf a = ld_afrag(arow + k0);
#pragma unroll
                for (int c = 0; c < 4; ++c) {
                    const __bf16* bp =
                        lds_w + (size_t)(c * 16 + nl) * LDS_STRIDE + k0 + g * 8;
                    acc[c] = wmma_bf16(a, ld_bfrag(bp), acc[c]);
                }
            }
        }
        // epilogue: h = tanh(acc + xp_t); store bf16 into hs[t]
        const float* xpt = xp + (size_t)t * B_DIM * H_DIM;
        __bf16*      ht  = hs + (size_t)t * B_DIM * H_DIM;
#pragma unroll
        for (int c = 0; c < 4; ++c) {
            int n = n0 + c * 16 + nl;
#pragma unroll
            for (int r = 0; r < 8; ++r) {
                int m = m0 + r + g * 8;
                float v = fast_tanh(acc[c][r] + xpt[(size_t)m * H_DIM + n]);
                ht[(size_t)m * H_DIM + n] = (__bf16)v;
            }
        }
        grid_barrier(bar, bar + 1, gridDim.x);
    }
}

// ---------------- FC head: out[t,b] = dot(hs[t,b,:], W_fc) + b_fc -----------

__global__ void fc_head(const __bf16* __restrict__ hs,
                        const float* __restrict__ wfc,
                        const float* __restrict__ bfc,
                        float* __restrict__ out)
{
    int lane = threadIdx.x & 31;
    int wave = threadIdx.x >> 5;
    int row  = blockIdx.x * 8 + wave;          // t*B + b, 0..32767
    const __bf16* h = hs + (size_t)row * H_DIM;
    float acc = 0.0f;
#pragma unroll
    for (int it = 0; it < 4; ++it) {
        int k = it * 256 + lane * 8;
        v8bf hv = *(const v8bf*)(h + k);
        v8f  wv = *(const v8f*)(wfc + k);
#pragma unroll
        for (int j = 0; j < 8; ++j) acc += (float)hv[j] * wv[j];
    }
#pragma unroll
    for (int off = 16; off > 0; off >>= 1) acc += __shfl_xor(acc, off, 32);
    if (lane == 0) out[row] = acc + bfc[0];
}

__global__ void init_bar(unsigned* bar) {
    if (threadIdx.x < 2) bar[threadIdx.x] = 0u;
}

// ---------------- host launcher ---------------------------------------------

extern "C" void kernel_launch(void* const* d_in, const int* in_sizes, int n_in,
                              void* d_out, int out_size, void* d_ws, size_t ws_size,
                              hipStream_t stream) {
    (void)in_sizes; (void)n_in; (void)out_size; (void)ws_size;

    const float* x    = (const float*)d_in[0];  // [T,B,I]
    const float* W_ih = (const float*)d_in[1];  // [H,I]
    const float* W_hh = (const float*)d_in[2];  // [H,H]
    const float* b_ih = (const float*)d_in[3];  // [H]
    const float* b_hh = (const float*)d_in[4];  // [H]
    const float* W_fc = (const float*)d_in[5];  // [O,H] = [H]
    const float* b_fc = (const float*)d_in[6];  // [O]

    // workspace layout
    char* ws = (char*)d_ws;
    const size_t XP_BYTES  = (size_t)T_DIM * B_DIM * H_DIM * 4;   // 128 MB fp32
    const size_t HS_BYTES  = (size_t)T_DIM * B_DIM * H_DIM * 2;   //  64 MB bf16
    const size_t XBF_BYTES = (size_t)T_DIM * B_DIM * I_DIM * 2;   //  32 MB bf16
    const size_t WIH_BYTES = (size_t)H_DIM * I_DIM * 2;
    const size_t WHH_BYTES = (size_t)H_DIM * H_DIM * 2;

    float*    xp     = (float*)ws;
    __bf16*   hs     = (__bf16*)(ws + XP_BYTES);
    __bf16*   xbf    = (__bf16*)(ws + XP_BYTES + HS_BYTES);
    __bf16*   wihbf  = (__bf16*)(ws + XP_BYTES + HS_BYTES + XBF_BYTES);
    __bf16*   whhbf  = (__bf16*)(ws + XP_BYTES + HS_BYTES + XBF_BYTES + WIH_BYTES);
    unsigned* bar    = (unsigned*)(ws + XP_BYTES + HS_BYTES + XBF_BYTES + WIH_BYTES + WHH_BYTES);

    // 1) precision conversion
    {
        int n4 = (T_DIM * B_DIM * I_DIM) / 4;
        cvt_f32_to_bf16<<<n4 / 256, 256, 0, stream>>>(x, xbf, n4);
        n4 = (H_DIM * I_DIM) / 4;
        cvt_f32_to_bf16<<<n4 / 256, 256, 0, stream>>>(W_ih, wihbf, n4);
        n4 = (H_DIM * H_DIM) / 4;
        cvt_f32_to_bf16<<<n4 / 256, 256, 0, stream>>>(W_hh, whhbf, n4);
    }
    init_bar<<<1, 32, 0, stream>>>(bar);

    // 2) input projection GEMM (32768 x 512 x 1024)
    xp_gemm<<<4096, 256, 0, stream>>>(xbf, wihbf, b_ih, b_hh, xp);

    // 3) persistent recurrence: 16 blocks, 256 steps, grid barrier per step
    size_t lds_bytes = (size_t)64 * LDS_STRIDE * sizeof(__bf16);  // ~129 KB
    rnn_steps<<<16, 256, lds_bytes, stream>>>(xp, whhbf, hs, bar);

    // 4) FC head over all timesteps
    fc_head<<<4096, 256, 0, stream>>>(hs, W_fc, b_fc, (float*)d_out);
}
